// InternLM3AttentionCustom_55937654063481
// MI455X (gfx1250) — matrix operand order repaired and checked by reading, not compiled, not run on hardware
//
#include <hip/hip_runtime.h>
#include <hip/hip_bf16.h>

// ---------------------------------------------------------------------------
// InternLM3 attention block for MI455X (gfx1250, wave32, WMMA).
// Pipeline: cvt->bf16, RoPE(Q), WMMA GEMM K=X*Wqk^T (+fused RoPE),
//           WMMA GEMM V=X*Wv^T (transposed store), flash attention (WMMA),
//           WMMA GEMM out = attn*Wo^T (fp32 store).
// ---------------------------------------------------------------------------

typedef __attribute__((ext_vector_type(16))) __bf16 v16bf;
typedef __attribute__((ext_vector_type(8)))  float  v8f;

union Frag16 {
    v16bf v;
    uint4 q[2];
};

__device__ __forceinline__ unsigned short f2bf(float f) {
    unsigned u = __float_as_uint(f);
    unsigned r = u + 0x7FFFu + ((u >> 16) & 1u);   // round-to-nearest-even
    return (unsigned short)(r >> 16);
}

__device__ __forceinline__ float bf2f(unsigned short h) {
    return __uint_as_float(((unsigned)h) << 16);
}

// ln(10000)/32 : inv_freq[d] = exp(-d * LOG_THETA_OVER_32), d in [0,32)
#define LOG_THETA_OVER_32 0.28782313662425572f
#define NEG_INF_F (-1e9f)

// ---------------------------------------------------------------------------
// fp32 -> bf16 bulk convert (vectorized: float4 in, 4x bf16 out)
// ---------------------------------------------------------------------------
__global__ void __launch_bounds__(256)
cvt_f32_bf16(const float* __restrict__ src, unsigned short* __restrict__ dst, int n4) {
    int i = blockIdx.x * blockDim.x + threadIdx.x;
    if (i >= n4) return;
    float4 f = reinterpret_cast<const float4*>(src)[i];
    uint2 p;
    p.x = (unsigned)f2bf(f.x) | ((unsigned)f2bf(f.y) << 16);
    p.y = (unsigned)f2bf(f.z) | ((unsigned)f2bf(f.w) << 16);
    reinterpret_cast<uint2*>(dst)[i] = p;
}

// ---------------------------------------------------------------------------
// RoPE on raw hidden -> Q (bf16). One thread per (pos, head, d<32) pair.
// ---------------------------------------------------------------------------
__global__ void __launch_bounds__(256)
rope_q_kernel(const float* __restrict__ hid, const long long* __restrict__ pos_ids,
              unsigned short* __restrict__ Qr, int S) {
    int idx = blockIdx.x * blockDim.x + threadIdx.x;
    if (idx >= S * 1024) return;               // S * 32 heads * 32 pairs
    int d    = idx & 31;
    int h    = (idx >> 5) & 31;
    int srow = idx >> 10;
    const float* x = hid + (size_t)srow * 2048 + h * 64;
    float lo = x[d], hi = x[d + 32];
    float p    = (float)pos_ids[srow];
    float invf = __expf(-(float)d * LOG_THETA_OVER_32);
    float ang  = p * invf;
    float c = cosf(ang), s = sinf(ang);
    unsigned short* y = Qr + (size_t)srow * 2048 + h * 64;
    y[d]      = f2bf(lo * c - hi * s);
    y[d + 32] = f2bf(hi * c + lo * s);
}

// ---------------------------------------------------------------------------
// Tiled bf16 WMMA GEMM: C[M,N] = A[M,K] * W[N,K]^T
// MODE 1: bf16 out + fused RoPE epilogue (K projection; wave N-tile == 1 head)
// MODE 2: bf16 out, transposed store out[col*M + row]  (V projection)
// MODE 3: fp32 out, plain store                        (O projection)
// Workgroup = 128 threads = 4 waves; each wave owns a 64x64 tile (16 accs).
// ---------------------------------------------------------------------------
template <int MODE>
__global__ void __launch_bounds__(128)
gemm_wmma(const unsigned short* __restrict__ A, const unsigned short* __restrict__ W,
          void* __restrict__ out, const long long* __restrict__ pos_ids,
          int M, int N, int K) {
    __shared__ unsigned short As[128][40];   // 128 rows x 32 halfs (+8 pad)
    __shared__ unsigned short Bs[128][40];

    const int tid   = threadIdx.x;
    const int lane  = tid & 31;
    const int wave  = tid >> 5;
    const int lhalf = lane & 15;
    const int lhi   = lane >> 4;
    const int m0 = blockIdx.y * 128;
    const int n0 = blockIdx.x * 128;
    const int wm = (wave >> 1) * 64;
    const int wn = (wave & 1) * 64;

    v8f acc[4][4];
    for (int a = 0; a < 4; ++a)
        for (int b = 0; b < 4; ++b)
            acc[a][b] = (v8f){0.f, 0.f, 0.f, 0.f, 0.f, 0.f, 0.f, 0.f};

    const unsigned short* arow = A + (size_t)(m0 + tid) * K;
    const unsigned short* brow = W + (size_t)(n0 + tid) * K;

    for (int k0 = 0; k0 < K; k0 += 32) {
        // ---- stage 128x32 A and B slabs into LDS (64B per thread each) ----
        {
            const uint4* ga = reinterpret_cast<const uint4*>(arow + k0);
            const uint4* gb = reinterpret_cast<const uint4*>(brow + k0);
            uint4* la = reinterpret_cast<uint4*>(&As[tid][0]);
            uint4* lb = reinterpret_cast<uint4*>(&Bs[tid][0]);
            la[0] = ga[0]; la[1] = ga[1]; la[2] = ga[2]; la[3] = ga[3];
            lb[0] = gb[0]; lb[1] = gb[1]; lb[2] = gb[2]; lb[3] = gb[3];
            if (k0 + 32 < K) {                       // prefetch next slab
                __builtin_prefetch(arow + k0 + 32, 0, 0);
                __builtin_prefetch(brow + k0 + 32, 0, 0);
            }
        }
        __syncthreads();

        // ---- B fragments: lane = col n (lhalf), 16 contiguous K halfs ----
        Frag16 bf[4];
        for (int b = 0; b < 4; ++b) {
            const uint4* p = reinterpret_cast<const uint4*>(&Bs[wn + b * 16 + lhalf][lhi * 16]);
            bf[b].q[0] = p[0];
            bf[b].q[1] = p[1];
        }
        // ---- A fragments + 16 WMMAs ----
        for (int a = 0; a < 4; ++a) {
            Frag16 af;
            af.q[0] = *reinterpret_cast<const uint4*>(&As[wm + a * 16 + lhalf][lhi * 16]);
            af.q[1] = *reinterpret_cast<const uint4*>(&As[wm + a * 16 + lhalf][lhi * 16 + 8]);
            for (int b = 0; b < 4; ++b)
                acc[a][b] = __builtin_amdgcn_wmma_f32_16x16x32_bf16(
                    false, af.v, false, bf[b].v, (short)0, acc[a][b], false, false);
        }
        __syncthreads();
    }

    // ---- epilogue ----
    unsigned short* ob = reinterpret_cast<unsigned short*>(out);
    float*          of = reinterpret_cast<float*>(out);

    if (MODE == 1) {
        // RoPE: wave col base (n0+wn) is 64-aligned => exactly one head.
        // dim d = b*16 + lhalf (<32) pairs with d+32 in frag b+2.
        for (int b = 0; b < 2; ++b) {
            int   d    = b * 16 + lhalf;
            float invf = __expf(-(float)d * LOG_THETA_OVER_32);
            for (int a = 0; a < 4; ++a) {
                for (int j = 0; j < 8; ++j) {
                    int   row = m0 + wm + a * 16 + j + 8 * lhi;
                    float p   = (float)pos_ids[row];
                    float ang = p * invf;
                    float c = cosf(ang), s = sinf(ang);
                    float lo = acc[a][b][j], hi = acc[a][b + 2][j];
                    int   col = n0 + wn + b * 16 + lhalf;
                    ob[(size_t)row * N + col]      = f2bf(lo * c - hi * s);
                    ob[(size_t)row * N + col + 32] = f2bf(hi * c + lo * s);
                }
            }
        }
    } else {
        for (int a = 0; a < 4; ++a)
            for (int b = 0; b < 4; ++b)
                for (int j = 0; j < 8; ++j) {
                    int   row = m0 + wm + a * 16 + j + 8 * lhi;
                    int   col = n0 + wn + b * 16 + lhalf;
                    float v   = acc[a][b][j];
                    if (MODE == 2)      ob[(size_t)col * M + row] = f2bf(v); // V^T
                    else if (MODE == 3) of[(size_t)row * N + col] = v;      // fp32
                    else                ob[(size_t)row * N + col] = f2bf(v);
                }
    }
}

// ---------------------------------------------------------------------------
// Flash attention, one wave per (16-query block, head).
// Q,K: bf16 [S][2048] row-major (post-RoPE).  Vt: bf16 [32*64][S] (V^T).
// Per 32-key block: 4 WMMAs for QK^T, online softmax, 4 WMMAs for PV.
// ---------------------------------------------------------------------------
__global__ void __launch_bounds__(32)
flash_attn(const unsigned short* __restrict__ Q, const unsigned short* __restrict__ Kk,
           const unsigned short* __restrict__ Vt, unsigned short* __restrict__ Oout,
           int S) {
    __shared__ unsigned short Pld[16][32];     // P tile staging (A-frag reshape)

    const int lane  = threadIdx.x & 31;
    const int lhalf = lane & 15;
    const int lhi   = lane >> 4;
    const int q0    = blockIdx.x * 16;
    const int hoff  = blockIdx.y * 64;

    // Q A-fragments for K-dim halves [0,32) and [32,64)
    Frag16 qf[2];
    {
        const unsigned short* qrow = Q + (size_t)(q0 + lhalf) * 2048 + hoff;
        qf[0].q[0] = *reinterpret_cast<const uint4*>(qrow + lhi * 8);
        qf[0].q[1] = *reinterpret_cast<const uint4*>(qrow + 16 + lhi * 8);
        qf[1].q[0] = *reinterpret_cast<const uint4*>(qrow + 32 + lhi * 8);
        qf[1].q[1] = *reinterpret_cast<const uint4*>(qrow + 48 + lhi * 8);
    }

    float mrow[8], lrow[8];
    v8f   o[4];
    for (int j = 0; j < 8; ++j) { mrow[j] = -1e30f; lrow[j] = 0.f; }
    for (int nd = 0; nd < 4; ++nd)
        o[nd] = (v8f){0.f, 0.f, 0.f, 0.f, 0.f, 0.f, 0.f, 0.f};

    for (int kb = 0; kb <= q0 + 15; kb += 32) {
        // ---- scores for two 16-key halves ----
        v8f sc[2];
        for (int t = 0; t < 2; ++t) {
            v8f accs = (v8f){0.f, 0.f, 0.f, 0.f, 0.f, 0.f, 0.f, 0.f};
            for (int kk = 0; kk < 2; ++kk) {
                Frag16 kf;
                const unsigned short* krow =
                    Kk + (size_t)(kb + t * 16 + lhalf) * 2048 + hoff + kk * 32;
                kf.q[0] = *reinterpret_cast<const uint4*>(krow + lhi * 8);
                kf.q[1] = *reinterpret_cast<const uint4*>(krow + 16 + lhi * 8);
                accs = __builtin_amdgcn_wmma_f32_16x16x32_bf16(
                    false, qf[kk].v, false, kf.v, (short)0, accs, false, false);
            }
            sc[t] = accs;
        }

        // ---- online softmax (row = q0 + j + 8*lhi, col across 16 lanes) ----
        float alpha[8];
        for (int j = 0; j < 8; ++j) {
            int   row = q0 + j + 8 * lhi;
            float s0 = sc[0][j] * 0.125f;
            float s1 = sc[1][j] * 0.125f;
            if (kb + lhalf > row)      s0 += NEG_INF_F;
            if (kb + 16 + lhalf > row) s1 += NEG_INF_F;
            float mx = fmaxf(s0, s1);
            for (int off = 1; off < 16; off <<= 1)
                mx = fmaxf(mx, __shfl_xor(mx, off, 16));
            float mnew = fmaxf(mrow[j], mx);
            alpha[j] = __expf(mrow[j] - mnew);
            float p0 = __expf(s0 - mnew);
            float p1 = __expf(s1 - mnew);
            float rs = p0 + p1;
            for (int off = 1; off < 16; off <<= 1)
                rs += __shfl_xor(rs, off, 16);
            lrow[j] = lrow[j] * alpha[j] + rs;
            mrow[j] = mnew;
            Pld[j + 8 * lhi][lhalf]      = f2bf(p0);
            Pld[j + 8 * lhi][16 + lhalf] = f2bf(p1);
        }
        for (int nd = 0; nd < 4; ++nd)
            for (int j = 0; j < 8; ++j)
                o[nd][j] *= alpha[j];

        __syncthreads();    // single wave: orders LDS stores before A-frag reads

        // ---- reshape P into an A-fragment, then PV ----
        Frag16 pf;
        pf.q[0] = *reinterpret_cast<const uint4*>(&Pld[lhalf][lhi * 8]);
        pf.q[1] = *reinterpret_cast<const uint4*>(&Pld[lhalf][16 + lhi * 8]);
        for (int nd = 0; nd < 4; ++nd) {
            Frag16 vf;
            const unsigned short* vrow =
                Vt + (size_t)(hoff + nd * 16 + lhalf) * S + kb;
            vf.q[0] = *reinterpret_cast<const uint4*>(vrow + lhi * 8);
            vf.q[1] = *reinterpret_cast<const uint4*>(vrow + 16 + lhi * 8);
            o[nd] = __builtin_amdgcn_wmma_f32_16x16x32_bf16(
                false, pf.v, false, vf.v, (short)0, o[nd], false, false);
        }
        __syncthreads();
    }

    // ---- finalize: divide by l, store bf16 [S][2048] head-major cols ----
    for (int nd = 0; nd < 4; ++nd)
        for (int j = 0; j < 8; ++j) {
            int row = q0 + j + 8 * lhi;
            int col = hoff + nd * 16 + lhalf;
            Oout[(size_t)row * 2048 + col] = f2bf(o[nd][j] / lrow[j]);
        }
}

// ---------------------------------------------------------------------------
extern "C" void kernel_launch(void* const* d_in, const int* in_sizes, int n_in,
                              void* d_out, int out_size, void* d_ws, size_t ws_size,
                              hipStream_t stream) {
    (void)in_sizes; (void)n_in; (void)out_size; (void)ws_size;
    const float*     hid = (const float*)d_in[0];
    const float*     qkw = (const float*)d_in[1];
    const float*     vw  = (const float*)d_in[2];
    const float*     ow  = (const float*)d_in[3];
    const long long* pos = (const long long*)d_in[4];
    float* out = (float*)d_out;

    const int    S = 2048, HID = 2048;
    const size_t E = (size_t)S * HID;          // 4 M elements
    unsigned short* ws   = (unsigned short*)d_ws;
    unsigned short* hidb = ws;                 // bf16 hidden
    unsigned short* qkwb = ws + E;             // bf16 qk_w
    unsigned short* vwb  = ws + 2 * E;         // bf16 v_w
    unsigned short* owb  = ws + 3 * E;         // bf16 o_w
    unsigned short* Kr   = ws + 4 * E;         // roped keys   [S][2048]
    unsigned short* Vt   = ws + 5 * E;         // values^T     [2048][S]
    unsigned short* Qr   = ws + 6 * E;         // roped queries[S][2048]
    unsigned short* At   = ws + 7 * E;         // attn output  [S][2048]

    const int n4 = (int)(E / 4);
    cvt_f32_bf16<<<n4 / 256, 256, 0, stream>>>(hid, hidb, n4);
    cvt_f32_bf16<<<n4 / 256, 256, 0, stream>>>(qkw, qkwb, n4);
    cvt_f32_bf16<<<n4 / 256, 256, 0, stream>>>(vw,  vwb,  n4);
    cvt_f32_bf16<<<n4 / 256, 256, 0, stream>>>(ow,  owb,  n4);
    rope_q_kernel<<<(S * 1024) / 256, 256, 0, stream>>>(hid, pos, Qr, S);

    dim3 gg(HID / 128, S / 128), gb(128);
    gemm_wmma<1><<<gg, gb, 0, stream>>>(hidb, qkwb, Kr, pos, S, HID, HID);
    gemm_wmma<2><<<gg, gb, 0, stream>>>(hidb, vwb,  Vt, nullptr, S, HID, HID);
    flash_attn<<<dim3(S / 16, 32), 32, 0, stream>>>(Qr, Kr, Vt, At, S);
    gemm_wmma<3><<<gg, gb, 0, stream>>>(At, owb, out, nullptr, S, HID, HID);
}
